// Block_53472342835790
// MI455X (gfx1250) — matrix operand order repaired and checked by reading, not compiled, not run on hardware
//
#include <hip/hip_runtime.h>
#include <math.h>

// ---------------------------------------------------------------------------
// Problem constants (from the reference)
// ---------------------------------------------------------------------------
#define M_      8
#define DOUT    20
#define SP      8000          // 20^3
#define NPOS    16000         // 2 * 8000
#define PADG    26            // padded spatial: 20 + 2*3
#define CIN_R   144           // real input channels to conv
#define CIN_P   160           // padded to 5 chunks of 32 for WMMA K-dim
#define OC_R    72
#define NTAP    343           // 7^3
#define ATILE   2560          // ushort elements per (tap, octile) A tile: 5*32*16

typedef __bf16 bf16_t;
typedef bf16_t  v16bf __attribute__((ext_vector_type(16)));
typedef float   v8f   __attribute__((ext_vector_type(8)));
typedef unsigned int v4u __attribute__((ext_vector_type(4)));
typedef unsigned int u32x4 __attribute__((ext_vector_type(4)));
typedef int i32x4 __attribute__((ext_vector_type(4)));
typedef int i32x8 __attribute__((ext_vector_type(8)));

union Frag { v16bf v; v4u u[2]; };

#if defined(__has_builtin)
#if __has_builtin(__builtin_amdgcn_tensor_load_to_lds) && \
    __has_builtin(__builtin_amdgcn_s_wait_tensorcnt)
#define USE_TDM 1
#endif
#endif
#ifndef USE_TDM
#define USE_TDM 0
#endif

__device__ inline unsigned short f2bf(float f) {
  unsigned u = __float_as_uint(f);
  u += 0x7FFFu + ((u >> 16) & 1u);       // round to nearest even
  return (unsigned short)(u >> 16);
}

#if USE_TDM
// Issue a TDM DMA of one contiguous A tile (ATILE bf16 elements) into LDS.
// D# built per CDNA5 ISA 8.3/8.4: count=1, type=2 ("image"), data_size=2B,
// 1-row tile of ATILE elements. shA is the only LDS object -> base offset 0.
__device__ inline void tdm_load_tile(const unsigned short* gsrc, unsigned lds_byte_off) {
  unsigned long long ga = (unsigned long long)(size_t)gsrc;
  u32x4 g0;
  g0[0] = 1u;                                               // count = 1 (valid user D#)
  g0[1] = lds_byte_off;                                     // lds_addr
  g0[2] = (unsigned)(ga & 0xFFFFFFFFu);                     // global_addr[31:0]
  g0[3] = (unsigned)((ga >> 32) & 0x01FFFFFFu) | (2u << 30);// global_addr[56:32] | type=2
  i32x8 g1;
  g1[0] = (int)(1u << 16);                                  // data_size = 1 -> 2 bytes
  g1[1] = (int)((ATILE & 0xFFFFu) << 16);                   // tensor_dim0[15:0] @ [63:48]
  g1[2] = (int)(1u << 16);                                  // tensor_dim0 hi=0, tensor_dim1 lo=1
  g1[3] = (int)((ATILE & 0xFFFFu) << 16);                   // tile_dim0 @ [127:112]
  g1[4] = 1;                                                // tile_dim1 = 1
  g1[5] = ATILE;                                            // tensor_dim0_stride
  g1[6] = 0;
  g1[7] = 0;
  i32x4 gz = {0, 0, 0, 0};
#if __clang_major__ >= 23
  i32x8 g4 = {0, 0, 0, 0, 0, 0, 0, 0};
  __builtin_amdgcn_tensor_load_to_lds(g0, g1, gz, gz, g4, 0);
#else
  __builtin_amdgcn_tensor_load_to_lds(g0, g1, gz, gz, 0);
#endif
}
#endif

// ---------------------------------------------------------------------------
// Device math: factorials, SU(2) CG, complex->real change of basis, real CG
// (direct port of the reference's numpy code, double precision)
// ---------------------------------------------------------------------------
__device__ double dfactorial(int n) {
  double r = 1.0;
  for (int i = 2; i <= n; ++i) r *= (double)i;
  return r;
}

__device__ double su2_cg(int j1, int j2, int j3, int m1, int m2, int m3) {
  if (m1 + m2 != m3) return 0.0;
  int dj = j1 - j2; if (dj < 0) dj = -dj;
  if (j3 < dj || j3 > j1 + j2) return 0.0;
  double pref = sqrt((2.0 * j3 + 1.0) * dfactorial(j3 + j1 - j2) * dfactorial(j3 - j1 + j2)
                     * dfactorial(j1 + j2 - j3) / dfactorial(j1 + j2 + j3 + 1));
  pref *= sqrt(dfactorial(j3 + m3) * dfactorial(j3 - m3) * dfactorial(j1 - m1)
               * dfactorial(j1 + m1) * dfactorial(j2 - m2) * dfactorial(j2 + m2));
  double s = 0.0;
  for (int k = 0; k <= j1 + j2 + j3; ++k) {
    int d0 = k, d1 = j1 + j2 - j3 - k, d2 = j1 - m1 - k, d3 = j2 + m2 - k;
    int d4 = j3 - j2 + m1 + k, d5 = j3 - j1 - m2 + k;
    if (d0 >= 0 && d1 >= 0 && d2 >= 0 && d3 >= 0 && d4 >= 0 && d5 >= 0) {
      double denom = dfactorial(d0) * dfactorial(d1) * dfactorial(d2)
                   * dfactorial(d3) * dfactorial(d4) * dfactorial(d5);
      s += ((k & 1) ? -1.0 : 1.0) / denom;
    }
  }
  return pref * s;
}

struct cplx { double re, im; };

__device__ cplx u_entry(int l, int row, int col) {
  const double is2 = 0.70710678118654752440;
  int mr = row - l, mc = col - l;
  cplx z; z.re = 0.0; z.im = 0.0;
  if (mr == 0) { if (mc == 0) z.re = 1.0; }
  else if (mr > 0) {
    int m = mr; double sgn = (m & 1) ? -1.0 : 1.0;
    if (mc == m) z.re = sgn * is2;
    else if (mc == -m) z.re = is2;
  } else {
    int m = -mr; double sgn = (m & 1) ? -1.0 : 1.0;
    if (mc == -m) z.im = is2;
    else if (mc == m) z.im = -sgn * is2;
  }
  return z;
}

// T[a,i,j] of _real_cg(l1,l2,l3): a over l3, i over l1, j over l2
__device__ void real_cg_entry(int l1, int l2, int l3, int a, int i, int j,
                              double* ore, double* oim) {
  double sre = 0.0, sim = 0.0;
  for (int c = 0; c < 2 * l3 + 1; ++c) {
    cplx u3 = u_entry(l3, a, c);
    if (u3.re == 0.0 && u3.im == 0.0) continue;
    for (int m = 0; m < 2 * l1 + 1; ++m) {
      cplx u1 = u_entry(l1, i, m);
      if (u1.re == 0.0 && u1.im == 0.0) continue;
      for (int n = 0; n < 2 * l2 + 1; ++n) {
        cplx u2 = u_entry(l2, j, n);
        if (u2.re == 0.0 && u2.im == 0.0) continue;
        double cg = su2_cg(l1, l2, l3, m - l1, n - l2, c - l3);
        if (cg == 0.0) continue;
        double pre = u1.re * u2.re - u1.im * u2.im;
        double pim = -(u1.re * u2.im + u1.im * u2.re);
        double qre = u3.re * pre - u3.im * pim;
        double qim = u3.re * pim + u3.im * pre;
        sre += qre * cg; sim += qim * cg;
      }
    }
  }
  *ore = sre; *oim = sim;
}

// Real spherical harmonics for l=0..4 at one point (port of _real_sph_all)
__device__ void sph_all(double x, double y, double z, double Y[5][9]) {
  const double PI_ = 3.14159265358979323846;
  double r = sqrt(x * x + y * y + z * z);
  const double eps = 1e-9;
  double ct = (r > eps) ? z / fmax(r, eps) : 1.0;
  double phi = atan2(y, x);
  double st = sqrt(fmax(1.0 - ct * ct, 0.0));
  double P[5][5];
  P[0][0] = 1.0;
  for (int m = 1; m <= 4; ++m) {
    double dfac = 1.0;
    for (int k = 1; k < 2 * m; k += 2) dfac *= (double)k;
    double sm = 1.0;
    for (int k = 0; k < m; ++k) sm *= st;
    P[m][m] = ((m & 1) ? -1.0 : 1.0) * dfac * sm;
  }
  for (int m = 0; m < 4; ++m) P[m + 1][m] = ct * (2.0 * m + 1.0) * P[m][m];
  for (int m = 0; m <= 4; ++m)
    for (int l = m + 2; l <= 4; ++l)
      P[l][m] = ((2.0 * l - 1.0) * ct * P[l - 1][m] - (l + m - 1.0) * P[l - 2][m]) / (double)(l - m);
  for (int l = 0; l <= 4; ++l) {
    for (int mm = 0; mm < 2 * l + 1; ++mm) Y[l][mm] = 0.0;
    for (int m = 0; m <= l; ++m) {
      double N = sqrt((2.0 * l + 1.0) / (4.0 * PI_) * dfactorial(l - m) / dfactorial(l + m));
      if (m == 0) Y[l][l] = N * P[l][0];
      else {
        Y[l][l + m] = sqrt(2.0) * N * P[l][m] * cos((double)m * phi);
        Y[l][l - m] = sqrt(2.0) * N * P[l][m] * sin((double)m * phi);
      }
    }
    if (l > 0 && !(r > eps))
      for (int mm = 0; mm < 2 * l + 1; ++mm) Y[l][mm] = 0.0;
  }
}

// ---------------------------------------------------------------------------
// Kernel: TPC[0] (9 floats) and TPC[2] (45 floats) into ws  (TPC[1] => irr1==0)
// ---------------------------------------------------------------------------
__global__ void compute_tpc(float* tpc) {
  __shared__ double sre[54], sim[54];
  __shared__ unsigned mR[2], mI[2];
  int tid = threadIdx.x;
  if (tid < 2) { mR[tid] = 0; mI[tid] = 0; }
  __syncthreads();
  for (int e = tid; e < 54; e += blockDim.x) {
    int L, a, r;
    if (e < 9) { L = 0; a = 0; r = e; } else { L = 2; a = (e - 9) / 9; r = (e - 9) % 9; }
    int i = r / 3, j = r % 3;
    double re, im;
    real_cg_entry(1, 1, L, a, i, j, &re, &im);
    sre[e] = re; sim[e] = im;
    int g = (e < 9) ? 0 : 1;
    atomicMax(&mR[g], __float_as_uint(fabsf((float)re)));
    atomicMax(&mI[g], __float_as_uint(fabsf((float)im)));
  }
  __syncthreads();
  for (int e = tid; e < 54; e += blockDim.x) {
    int g = (e < 9) ? 0 : 1;
    tpc[e] = (float)((mI[g] > mR[g]) ? sim[e] : sre[e]);
  }
}

// ---------------------------------------------------------------------------
// Kernel: K[72][144][343] from weights + device-generated steerable basis.
// One block per (lo, li) pair.
// ---------------------------------------------------------------------------
__global__ __launch_bounds__(256) void compute_K(
    const float* w00, const float* w01, const float* w02,
    const float* w10, const float* w11, const float* w12,
    const float* w20, const float* w21, const float* w22,
    float* __restrict__ Kmat) {
  const int bx = blockIdx.x;
  const int lo = bx / 3, li = bx % 3;
  const float* w;
  switch (bx) {
    case 0: w = w00; break; case 1: w = w01; break; case 2: w = w02; break;
    case 3: w = w10; break; case 4: w = w11; break; case 5: w = w12; break;
    case 6: w = w20; break; case 7: w = w21; break; default: w = w22; break;
  }
  const int lmin = (lo > li) ? (lo - li) : (li - lo);
  const int lminc = (lo < li) ? lo : li;
  const int nl = 2 * lminc + 1;
  const int A = 2 * lo + 1, I = 2 * li + 1;
  const int rowOff = (lo == 0) ? 0 : ((lo == 1) ? 8 : 32);
  const int colOff = (li == 0) ? 0 : ((li == 1) ? 16 : 64);

  __shared__ float sT[625];
  __shared__ float sTim[625];
  __shared__ float sY[25 * NTAP];
  __shared__ float sRad[3 * NTAP];
  __shared__ float sNorm[15];
  __shared__ float sInv[15];
  __shared__ unsigned mR[5], mI[5];
  __shared__ int tOff[6], yOff[6];

  const int tid = threadIdx.x, nt = blockDim.x;
  if (tid == 0) {
    int to = 0, yo = 0;
    for (int k = 0; k < nl; ++k) {
      tOff[k] = to; yOff[k] = yo;
      int L = lmin + k; to += A * I * (2 * L + 1); yo += 2 * L + 1;
    }
    tOff[nl] = to; yOff[nl] = yo;
  }
  if (tid < 5) { mR[tid] = 0; mI[tid] = 0; }
  if (tid < 15) sNorm[tid] = 0.f;
  __syncthreads();

  // ---- real CG tensors T[a,i,m] per l (with the reference's real/imag pick)
  const int totT = tOff[nl];
  for (int e = tid; e < totT; e += nt) {
    int k = 0; while (k + 1 < nl && tOff[k + 1] <= e) ++k;
    int r = e - tOff[k]; int L = lmin + k; int w21 = 2 * L + 1;
    int a = r / (I * w21); int rem = r % (I * w21); int i = rem / w21; int m = rem % w21;
    double re, im;
    real_cg_entry(li, L, lo, a, i, m, &re, &im);
    sT[e] = (float)re; sTim[e] = (float)im;
    atomicMax(&mR[k], __float_as_uint(fabsf((float)re)));
    atomicMax(&mI[k], __float_as_uint(fabsf((float)im)));
  }
  __syncthreads();
  for (int e = tid; e < totT; e += nt) {
    int k = 0; while (k + 1 < nl && tOff[k + 1] <= e) ++k;
    if (mI[k] > mR[k]) sT[e] = sTim[e];
  }

  // ---- spherical harmonics and radial Gaussians over the 7^3 grid
  for (int tap = tid; tap < NTAP; tap += nt) {
    int td = tap / 49, th = (tap / 7) % 7, tw = tap % 7;
    double x = (double)(td - 3), y = (double)(th - 3), z = (double)(tw - 3);
    double Yl[5][9];
    sph_all(x, y, z, Yl);
    for (int k = 0; k < nl; ++k) {
      int L = lmin + k;
      for (int m = 0; m < 2 * L + 1; ++m) sY[(yOff[k] + m) * NTAP + tap] = (float)Yl[L][m];
    }
    double r = sqrt(x * x + y * y + z * z);
    for (int j = 0; j < 3; ++j) {
      double c = 1.5 * (double)j;
      double g = exp(-0.5 * (r - c) * (r - c));
      sRad[j * NTAP + tap] = (r <= 3.5) ? (float)g : 0.f;
    }
  }
  __syncthreads();

  // ---- Frobenius norms of each (l_idx, j) basis block
  const int perK = A * I * NTAP;
  for (int e = tid; e < nl * perK; e += nt) {
    int k = e / perK; int r = e % perK; int ai = r / NTAP; int tap = r % NTAP;
    int L = lmin + k; int w21 = 2 * L + 1;
    const float* Trow = &sT[tOff[k] + ai * w21];
    float ang = 0.f;
    for (int m = 0; m < w21; ++m) ang += Trow[m] * sY[(yOff[k] + m) * NTAP + tap];
    for (int j = 0; j < 3; ++j) {
      float v = ang * sRad[j * NTAP + tap];
      atomicAdd(&sNorm[k * 3 + j], v * v);
    }
  }
  __syncthreads();
  if (tid < nl * 3) {
    float n = sqrtf(sNorm[tid]);
    sInv[tid] = (n > 1e-12f) ? (1.f / n) : 1.f;
  }
  __syncthreads();

  // ---- contract weights with the normalized basis -> K entries
  const int rows = 8 * A, cols = 16 * I;
  const long long totK = (long long)rows * cols * NTAP;
  for (long long e = tid; e < totK; e += nt) {
    int tap = (int)(e % NTAP); long long t2 = e / NTAP;
    int colL = (int)(t2 % cols); int rowL = (int)(t2 / cols);
    int u = rowL / A, a = rowL % A, vch = colL / I, ich = colL % I;
    float sum = 0.f;
    for (int k = 0; k < nl; ++k) {
      int L = lmin + k; int w21 = 2 * L + 1;
      const float* Trow = &sT[tOff[k] + (a * I + ich) * w21];
      float ang = 0.f;
      for (int m = 0; m < w21; ++m) ang += Trow[m] * sY[(yOff[k] + m) * NTAP + tap];
      for (int j = 0; j < 3; ++j) {
        float wv = w[((u * 16 + vch) * nl + k) * 3 + j];
        sum += wv * sInv[k * 3 + j] * ang * sRad[j * NTAP + tap];
      }
    }
    Kmat[((size_t)(rowOff + rowL) * CIN_R + (colOff + colL)) * NTAP + tap] = sum;
  }
}

// ---------------------------------------------------------------------------
// Kernel: pack K into WMMA A-fragment order (bf16), pad OC 72->80, IC 144->160
// layout: [tap][oc_tile 5][chunk 5][lane 32][16 bf16]
// ---------------------------------------------------------------------------
__global__ void pack_A(const float* __restrict__ Kmat, unsigned short* __restrict__ ap) {
  int g = blockIdx.x * blockDim.x + threadIdx.x;
  const int tot = NTAP * 5 * 5 * 32;
  if (g >= tot) return;
  int lane = g & 31;
  int c = (g >> 5) % 5;
  int oct = (g / 160) % 5;
  int tap = g / 800;
  unsigned short* o = ap + (size_t)g * 16;
  int m = (lane < 16) ? lane : (lane - 16);
  int oc = oct * 16 + m;
  for (int vv = 0; vv < 16; ++vv) {
    int kk;
    if (lane < 16) kk = vv;        // K = 0..15
    else           kk = vv + 16;   // K = 16..31
    int ic = c * 32 + kk;
    float val = (oc < OC_R && ic < CIN_R) ? Kmat[((size_t)oc * CIN_R + ic) * NTAP + tap] : 0.f;
    o[vv] = f2bf(val);
  }
}

// ---------------------------------------------------------------------------
// Kernel: build padded, channel-last bf16 xin: [b][26][26][26][160]
// channels: [s(8) | irr0(8) | v(24) | irr1=0(24) | q(40) | irr2(40) | pad(16)]
// ---------------------------------------------------------------------------
__global__ void build_xin(const float* __restrict__ x, const float* __restrict__ tpc,
                          unsigned short* __restrict__ xin) {
  int idx = blockIdx.x * blockDim.x + threadIdx.x;
  const int tot = 2 * PADG * PADG * PADG;
  if (idx >= tot) return;
  int xp = idx % PADG; int t = idx / PADG;
  int yp = t % PADG; t /= PADG;
  int zp = t % PADG; int b = t / PADG;
  unsigned short* dst = xin + (size_t)idx * CIN_P;
  bool inside = (zp >= 3 && zp < 23 && yp >= 3 && yp < 23 && xp >= 3 && xp < 23);
  if (!inside) {
    for (int c = 0; c < CIN_P; ++c) dst[c] = 0;
    return;
  }
  int d = zp - 3, h = yp - 3, wv = xp - 3;
  int sp = d * 400 + h * 20 + wv;
  const float* xb = x + (size_t)b * OC_R * SP;
  float s[8], v[24], q[40];
  for (int u = 0; u < 8; ++u)  s[u] = xb[(size_t)u * SP + sp];
  for (int c = 0; c < 24; ++c) v[c] = xb[(size_t)(8 + c) * SP + sp];
  for (int c = 0; c < 40; ++c) q[c] = xb[(size_t)(32 + c) * SP + sp];
  for (int u = 0; u < 8; ++u) dst[u] = f2bf(s[u]);
  for (int u = 0; u < 8; ++u) {                       // irr0 = TPC0_ij v_i v_j
    float acc = 0.f;
    for (int i = 0; i < 3; ++i)
      for (int j = 0; j < 3; ++j)
        acc += tpc[i * 3 + j] * v[u * 3 + i] * v[u * 3 + j];
    dst[8 + u] = f2bf(acc);
  }
  for (int c = 0; c < 24; ++c) dst[16 + c] = f2bf(v[c]);
  for (int c = 0; c < 24; ++c) dst[40 + c] = 0;       // irr1 == 0 (antisymmetric CG)
  for (int c = 0; c < 40; ++c) dst[64 + c] = f2bf(q[c]);
  for (int u = 0; u < 8; ++u)
    for (int a = 0; a < 5; ++a) {                     // irr2 = TPC2_aij v_i v_j
      float acc = 0.f;
      for (int i = 0; i < 3; ++i)
        for (int j = 0; j < 3; ++j)
          acc += tpc[9 + a * 9 + i * 3 + j] * v[u * 3 + i] * v[u * 3 + j];
      dst[104 + u * 5 + a] = f2bf(acc);
    }
  for (int c = 144; c < CIN_P; ++c) dst[c] = 0;
}

// ---------------------------------------------------------------------------
// Kernel: implicit-GEMM 3D conv via v_wmma_f32_16x16x32_bf16.
// Wave tile: 16 oc x 64 positions (4 accumulators, A-frag reused 4x).
// Per-block A tile (5 KB/tap) staged in LDS, double-buffered, fetched by the
// Tensor Data Mover (async DMA, TENSORcnt) when available.
// grid = (63 pos-blocks, 5 oc-tiles), block = 128 threads = 4 waves.
// ---------------------------------------------------------------------------
__global__ __launch_bounds__(128) void conv_wmma(const unsigned short* __restrict__ xin,
                                                 const unsigned short* __restrict__ apack,
                                                 float* __restrict__ y) {
  __shared__ __align__(16) unsigned short shA[2][ATILE];
  const int tid = threadIdx.x;
  const int wave = tid >> 5;
  const int lane = tid & 31;
  const int octile = blockIdx.y;
  const int posBase = blockIdx.x * 256 + wave * 64;
  const int n = lane & 15;
  const int khalf = lane >> 4;

  int bAddr[4], spos[4], bb[4];
  bool valid[4];
#pragma unroll
  for (int q = 0; q < 4; ++q) {
    int p = posBase + q * 16 + n;
    valid[q] = (p < NPOS);
    int pc = valid[q] ? p : 0;
    int b = pc / SP; int r = pc % SP;
    int d = r / 400, h = (r / 20) % 20, wv = r % 20;
    spos[q] = r; bb[q] = b;
    bAddr[q] = ((b * PADG + d) * PADG + h) * PADG + wv;  // element index / CIN_P
  }

  v8f acc[4];
#pragma unroll
  for (int q = 0; q < 4; ++q)
#pragma unroll
    for (int r = 0; r < 8; ++r) acc[q][r] = 0.f;

  // ---- prologue: stage A tile for tap 0 into buffer 0
#if USE_TDM
  if (wave == 0) {
    tdm_load_tile(apack + (size_t)octile * ATILE, 0u);
    __builtin_amdgcn_s_wait_tensorcnt(0);
  }
#else
  for (int i = tid; i < ATILE / 8; i += 128)
    reinterpret_cast<v4u*>(&shA[0][0])[i] =
        reinterpret_cast<const v4u*>(apack + (size_t)octile * ATILE)[i];
#endif
  __syncthreads();

  for (int tap = 0; tap < NTAP; ++tap) {
    const int buf = tap & 1;
    const int td = tap / 49, th = (tap / 7) % 7, tw = tap % 7;
    const int tapOff = td * (PADG * PADG) + th * PADG + tw;

    // ---- stage next tap's A tile into the other buffer (async via TDM)
    if (tap + 1 < NTAP) {
      const unsigned short* nsrc = apack + ((size_t)(tap + 1) * 5 + octile) * ATILE;
#if USE_TDM
      if (wave == 0) tdm_load_tile(nsrc, (unsigned)((buf ^ 1) * (ATILE * 2)));
#else
      for (int i = tid; i < ATILE / 8; i += 128)
        reinterpret_cast<v4u*>(&shA[buf ^ 1][0])[i] =
            reinterpret_cast<const v4u*>(nsrc)[i];
#endif
      const int ntap = tap + 1;
      const int ntapOff = (ntap / 49) * (PADG * PADG) + ((ntap / 7) % 7) * PADG + (ntap % 7);
      __builtin_prefetch(xin + (size_t)(bAddr[0] + ntapOff) * CIN_P, 0, 0);
    }

    // ---- 5 ic-chunks x 4 position accumulators of WMMA
#pragma unroll
    for (int c = 0; c < 5; ++c) {
      Frag A;
      const v4u* aptr = reinterpret_cast<const v4u*>(&shA[buf][(c * 32 + lane) * 16]);
      A.u[0] = aptr[0];
      A.u[1] = aptr[1];
#pragma unroll
      for (int q = 0; q < 4; ++q) {
        Frag B;
        const v4u* bptr = reinterpret_cast<const v4u*>(
            xin + (size_t)(bAddr[q] + tapOff) * CIN_P + c * 32 + khalf * 16);
        B.u[0] = bptr[0];
        B.u[1] = bptr[1];
        acc[q] = __builtin_amdgcn_wmma_f32_16x16x32_bf16(
            false, A.v, false, B.v, (short)0, acc[q], false, false);
      }
    }

#if USE_TDM
    if (wave == 0 && tap + 1 < NTAP) __builtin_amdgcn_s_wait_tensorcnt(0);
#endif
    __syncthreads();
  }

#pragma unroll
  for (int q = 0; q < 4; ++q) {
    if (!valid[q]) continue;
#pragma unroll
    for (int r = 0; r < 8; ++r) {
      int oc = octile * 16 + r + (khalf ? 8 : 0);
      if (oc < OC_R)
        y[((size_t)(bb[q] * OC_R + oc)) * SP + spos[q]] = acc[q][r];
    }
  }
}

// ---------------------------------------------------------------------------
// Epilogue: irrep-norm batchnorm statistics + scale/bias/relu
// ---------------------------------------------------------------------------
__device__ inline int bn_group(int ch) {
  if (ch < 8) return ch;
  if (ch < 32) return 8 + (ch - 8) / 3;
  return 16 + (ch - 32) / 5;
}

__global__ void bn_init(float* n2) {
  int t = threadIdx.x;
  if (t < 24) n2[t] = 0.f;
}

__global__ void bn_reduce(const float* __restrict__ y, float* __restrict__ n2) {
  __shared__ float sh[24];
  int tid = threadIdx.x;
  if (tid < 24) sh[tid] = 0.f;
  __syncthreads();
  const int tot = 2 * OC_R * SP;
  for (int i = blockIdx.x * blockDim.x + tid; i < tot; i += gridDim.x * blockDim.x) {
    int ch = (i / SP) % OC_R;
    float v = y[i];
    atomicAdd(&sh[bn_group(ch)], v * v);
  }
  __syncthreads();
  if (tid < 24) atomicAdd(&n2[tid], sh[tid]);
}

__global__ void bn_apply(const float* __restrict__ y, const float* __restrict__ n2,
                         const float* __restrict__ gamma, const float* __restrict__ bias,
                         float* __restrict__ out) {
  int i = blockIdx.x * blockDim.x + threadIdx.x;
  const int tot = 2 * OC_R * SP;
  if (i >= tot) return;
  int ch = (i / SP) % OC_R;
  int g = bn_group(ch);
  float scale = gamma[g] * rsqrtf(n2[g] / 16000.f + 1e-5f);
  float v = y[i] * scale;
  if (ch < 8) v = fmaxf(v + bias[ch], 0.f);
  out[i] = v;
}

// ---------------------------------------------------------------------------
// Launcher
// ---------------------------------------------------------------------------
extern "C" void kernel_launch(void* const* d_in, const int* in_sizes, int n_in,
                              void* d_out, int out_size, void* d_ws, size_t ws_size,
                              hipStream_t stream) {
  (void)in_sizes; (void)n_in; (void)out_size; (void)ws_size;
  const float* x     = (const float*)d_in[0];
  const float* w[9];
  for (int i = 0; i < 9; ++i) w[i] = (const float*)d_in[1 + i];
  const float* gamma = (const float*)d_in[10];
  const float* bias  = (const float*)d_in[11];
  float* out = (float*)d_out;

  // workspace layout (bytes)
  const size_t XIN_OFF = 0;                                     // bf16 [2][26^3][160]
  const size_t XIN_BYT = (size_t)2 * PADG * PADG * PADG * CIN_P * 2;
  const size_t K_OFF   = XIN_OFF + XIN_BYT;                     // f32 [72][144][343]
  const size_t K_BYT   = (size_t)OC_R * CIN_R * NTAP * 4;
  const size_t AP_OFF  = K_OFF + K_BYT;                         // bf16 packed A-frags
  const size_t AP_BYT  = (size_t)NTAP * 5 * ATILE * 2;
  const size_t Y_OFF   = AP_OFF + AP_BYT;                       // f32 [2][72][8000]
  const size_t Y_BYT   = (size_t)2 * OC_R * SP * 4;
  const size_t N2_OFF  = Y_OFF + Y_BYT;                         // f32 [24]
  const size_t N2_BYT  = 128;
  const size_t TPC_OFF = N2_OFF + N2_BYT;                       // f32 [54]

  char* ws = (char*)d_ws;
  unsigned short* ws_xin = (unsigned short*)(ws + XIN_OFF);
  float*          ws_K   = (float*)(ws + K_OFF);
  unsigned short* ws_ap  = (unsigned short*)(ws + AP_OFF);
  float*          ws_y   = (float*)(ws + Y_OFF);
  float*          ws_n2  = (float*)(ws + N2_OFF);
  float*          ws_tpc = (float*)(ws + TPC_OFF);

  // 1) constants (CG tensors) on device
  compute_tpc<<<1, 64, 0, stream>>>(ws_tpc);

  // 2) assemble conv kernel K from weights x steerable basis
  compute_K<<<9, 256, 0, stream>>>(w[0], w[1], w[2], w[3], w[4], w[5], w[6], w[7], w[8], ws_K);

  // 3) pack A-fragments (bf16, WMMA lane order)
  {
    int tot = NTAP * 5 * 5 * 32;
    pack_A<<<(tot + 255) / 256, 256, 0, stream>>>(ws_K, ws_ap);
  }

  // 4) build padded channel-last bf16 input
  {
    int tot = 2 * PADG * PADG * PADG;
    build_xin<<<(tot + 255) / 256, 256, 0, stream>>>(x, ws_tpc, ws_xin);
  }

  // 5) implicit-GEMM conv (WMMA + TDM double-buffered A)
  {
    dim3 grid(63, 5);
    conv_wmma<<<grid, 128, 0, stream>>>(ws_xin, ws_ap, ws_y);
  }

  // 6) batchnorm stats + apply
  bn_init<<<1, 32, 0, stream>>>(ws_n2);
  bn_reduce<<<512, 256, 0, stream>>>(ws_y, ws_n2);
  {
    int tot = 2 * OC_R * SP;
    bn_apply<<<(tot + 255) / 256, 256, 0, stream>>>(ws_y, ws_n2, gamma, bias, out);
  }
}
